// CrossCondGPT2AC_42657615184227
// MI455X (gfx1250) — matrix operand order repaired and checked by reading, not compiled, not run on hardware
//
#include <hip/hip_runtime.h>

typedef __attribute__((ext_vector_type(4)))  __bf16 v4bf;
typedef __attribute__((ext_vector_type(8)))  __bf16 v8bf;
typedef __attribute__((ext_vector_type(16))) __bf16 v16bf;
typedef __attribute__((ext_vector_type(8)))  float  v8f;

union FragBF { v16bf v; v8bf h[2]; };

#define WMMA_BF16(A_, B_, C_) \
  __builtin_amdgcn_wmma_f32_16x16x32_bf16(false, (A_), false, (B_), (short)0, (C_), false, false)

// ---- constants ------------------------------------------------------------
#define CB   2      // batch
#define CT   256    // t
#define CC   768    // C
#define CH   12     // heads
#define CD   64     // head dim
#define CTT  768    // T = 3t
#define CV   512    // vocab
#define CR   1536   // B*T rows

// ---- fragment loaders -----------------------------------------------------
// A-operand (16x32 bf16, MxK): lane l -> row (l&15); lane-half selects K
// interleave: lo = K[ko..ko+7], hi = K[16+ko..16+ko+7], ko = (l>>4)*8.
__device__ inline v16bf ld_fragA(const __bf16* tile, int pitch) {
  int l = threadIdx.x & 31;
  int r = l & 15;
  int ko = (l >> 4) << 3;
  FragBF f;
  f.h[0] = *(const v8bf*)(tile + r * pitch + ko);
  f.h[1] = *(const v8bf*)(tile + r * pitch + ko + 16);
  return f.v;
}
// B-operand (32x16 bf16, KxN) sourced from a row-major [N][K] tile:
// lane l -> column N=(l&15); lo = K[ko..ko+7], hi = K[16+ko..16+ko+7], ko = (l>>4)*8.
__device__ inline v16bf ld_fragB(const __bf16* tile, int pitch) {
  int l = threadIdx.x & 31;
  int r = l & 15;
  int ko = (l >> 4) << 3;
  FragBF f;
  f.h[0] = *(const v8bf*)(tile + r * pitch + ko);
  f.h[1] = *(const v8bf*)(tile + r * pitch + ko + 16);
  return f.v;
}

__device__ inline v8f vzero8() {
  v8f z;
#pragma unroll
  for (int i = 0; i < 8; ++i) z[i] = 0.f;
  return z;
}

// ---- embedding ------------------------------------------------------------
__global__ __launch_bounds__(256) void embed_kernel(
    const int* __restrict__ idx_up, const int* __restrict__ idx_down,
    const float* __restrict__ cond, const float* __restrict__ tok_up,
    const float* __restrict__ tok_down, const float* __restrict__ pos,
    const float* __restrict__ cond_w, const float* __restrict__ cond_b,
    float* __restrict__ x) {
  __shared__ float sc[438];
  const int row = blockIdx.x;           // 0..1535
  const int b = row / CTT, i = row % CTT;
  float* xr = x + (size_t)row * CC;
  if (i < CT) {
    const float* cr = cond + (size_t)(b * CT + i) * 438;
    for (int m = threadIdx.x; m < 438; m += 256) sc[m] = cr[m];
    __syncthreads();
    for (int c = threadIdx.x; c < CC; c += 256) {
      const float* wr = cond_w + (size_t)c * 438;
      float acc = cond_b[c];
      for (int m = 0; m < 438; ++m) acc = fmaf(sc[m], wr[m], acc);
      xr[c] = acc + pos[(size_t)i * CC + c];
    }
  } else if (i < 2 * CT) {
    const float* tr = tok_up + (size_t)idx_up[b * CT + (i - CT)] * CC;
    for (int c = threadIdx.x; c < CC; c += 256)
      xr[c] = tr[c] + pos[(size_t)i * CC + c];
  } else {
    const float* tr = tok_down + (size_t)idx_down[b * CT + (i - 2 * CT)] * CC;
    for (int c = threadIdx.x; c < CC; c += 256)
      xr[c] = tr[c] + pos[(size_t)i * CC + c];
  }
}

// ---- layernorm (fp32 in -> bf16 out) --------------------------------------
__global__ __launch_bounds__(256) void ln_kernel(
    const float* __restrict__ x, const float* __restrict__ w,
    const float* __restrict__ bias, __bf16* __restrict__ out) {
  __shared__ float red[256];
  const int row = blockIdx.x;
  const int tid = threadIdx.x;
  const float* xr = x + (size_t)row * CC;
  float v0 = xr[tid], v1 = xr[tid + 256], v2 = xr[tid + 512];
  red[tid] = v0 + v1 + v2;
  __syncthreads();
  for (int st = 128; st > 0; st >>= 1) {
    if (tid < st) red[tid] += red[tid + st];
    __syncthreads();
  }
  const float mu = red[0] * (1.f / (float)CC);
  __syncthreads();
  const float d0 = v0 - mu, d1 = v1 - mu, d2 = v2 - mu;
  red[tid] = d0 * d0 + d1 * d1 + d2 * d2;
  __syncthreads();
  for (int st = 128; st > 0; st >>= 1) {
    if (tid < st) red[tid] += red[tid + st];
    __syncthreads();
  }
  const float rstd = rsqrtf(red[0] * (1.f / (float)CC) + 1e-5f);
  __bf16* orow = out + (size_t)row * CC;
  orow[tid]       = (__bf16)(d0 * rstd * w[tid]       + bias[tid]);
  orow[tid + 256] = (__bf16)(d1 * rstd * w[tid + 256] + bias[tid + 256]);
  orow[tid + 512] = (__bf16)(d2 * rstd * w[tid + 512] + bias[tid + 512]);
}

// ---- generic WMMA GEMM: out = A_bf16 @ W^T (+bias)(+res)(gelu) ------------
// A: [rows x K] bf16 row-major; W: [N x K] fp32 row-major.
// 256 threads = 8 waves; 128x128 tile; wave = 32 rows x 64 cols.
// k-step 64 (16 WMMA per barrier), double-buffered LDS, L2 prefetch ahead.
#define GP 72  // LDS row pitch (bf16 elems), conflict-avoiding pad

__device__ inline void stage_tile(__bf16* __restrict__ sAb, __bf16* __restrict__ sBb,
                                  const __bf16* __restrict__ A,
                                  const float* __restrict__ W,
                                  int lda, int K, int rowBase, int colBase,
                                  int kk, int tid) {
  // A tile: 128x64 bf16 (1024 x 16B chunks)
#pragma unroll
  for (int i = 0; i < 4; ++i) {
    int idx = tid + i * 256;
    int r = idx >> 3;
    int c = (idx & 7) << 3;
    uint4 d = *(const uint4*)(A + (size_t)(rowBase + r) * lda + kk + c);
    *(uint4*)(&sAb[r * GP + c]) = d;
  }
  // W tile: 128x64 fp32 -> bf16 (2048 x float4 chunks)
#pragma unroll
  for (int i = 0; i < 8; ++i) {
    int idx = tid + i * 256;
    int n = idx >> 4;
    int c = (idx & 15) << 2;
    float4 wv = *(const float4*)(W + (size_t)(colBase + n) * K + kk + c);
    v4bf o;
    o.x = (__bf16)wv.x; o.y = (__bf16)wv.y;
    o.z = (__bf16)wv.z; o.w = (__bf16)wv.w;
    *(v4bf*)(&sBb[n * GP + c]) = o;
  }
}

template <bool HB, bool HR, bool HG, bool OF, bool OB>
__global__ __launch_bounds__(256) void gemm_kernel(
    const __bf16* __restrict__ A, int lda, long long bsA,
    const float* __restrict__ W, const float* __restrict__ bias,
    float* __restrict__ outF, __bf16* __restrict__ outB, int ldo, long long bsO,
    const float* __restrict__ res, int K) {
  __shared__ __bf16 sA[2][128 * GP];
  __shared__ __bf16 sB[2][128 * GP];
  const int tid = threadIdx.x;
  const int wid = tid >> 5;
  const int wm = wid & 3;   // 4 row groups of 32
  const int wn = wid >> 2;  // 2 col groups of 64
  const int z = blockIdx.z;
  const int rowBase = blockIdx.y * 128;
  const int colBase = blockIdx.x * 128;
  A += (size_t)z * bsA;

  v8f acc[2][4];
#pragma unroll
  for (int m = 0; m < 2; ++m)
#pragma unroll
    for (int n = 0; n < 4; ++n) acc[m][n] = vzero8();

  stage_tile(sA[0], sB[0], A, W, lda, K, rowBase, colBase, 0, tid);
  __syncthreads();

  int buf = 0;
  for (int kk = 0; kk < K; kk += 64) {
    const int nxt = buf ^ 1;
    if (kk + 64 < K)
      stage_tile(sA[nxt], sB[nxt], A, W, lda, K, rowBase, colBase, kk + 64, tid);
    if (kk + 128 < K) {  // L2 prefetch two tiles ahead (global_prefetch_b8)
      __builtin_prefetch(A + (size_t)(rowBase + (tid >> 1)) * lda + kk + 128 + (tid & 1) * 32, 0, 1);
      __builtin_prefetch(W + (size_t)(colBase + (tid >> 1)) * K + kk + 128 + (tid & 1) * 32, 0, 1);
    }
#pragma unroll
    for (int ks = 0; ks < 64; ks += 32) {
      v16bf afr0 = ld_fragA(&sA[buf][(wm * 32) * GP + ks], GP);
      v16bf afr1 = ld_fragA(&sA[buf][(wm * 32 + 16) * GP + ks], GP);
#pragma unroll
      for (int n = 0; n < 4; ++n) {
        v16bf bfr = ld_fragB(&sB[buf][(wn * 64 + n * 16) * GP + ks], GP);
        acc[0][n] = WMMA_BF16(afr0, bfr, acc[0][n]);
        acc[1][n] = WMMA_BF16(afr1, bfr, acc[1][n]);
      }
    }
    __syncthreads();
    buf = nxt;
  }

  const int l = tid & 31;
  const int lr = l & 15;
  const int lh = l >> 4;
#pragma unroll
  for (int m = 0; m < 2; ++m)
#pragma unroll
    for (int n = 0; n < 4; ++n) {
      const int col = colBase + wn * 64 + n * 16 + lr;
      const float bval = HB ? bias[col] : 0.f;
#pragma unroll
      for (int v = 0; v < 8; ++v) {
        const int row = rowBase + wm * 32 + m * 16 + v + (lh << 3);
        const size_t oidx = (size_t)z * bsO + (size_t)row * ldo + col;
        float val = acc[m][n][v] + bval;
        if (HR) val += res[(size_t)row * ldo + col];
        if (HG) val = 0.5f * val * (1.f + erff(val * 0.70710678f));
        if (OF) outF[oidx] = val;
        if (OB) outB[oidx] = (__bf16)val;
      }
    }
}

// ---- flash attention ------------------------------------------------------
// qkv: [B*T, 3C] fp32 (Q at h*64, K at C+h*64, V at 2C+h*64).
// grid (T/64, H, B), 128 threads = 4 waves, wave owns 16 query rows.
#define AP 72  // LDS pitch

__global__ __launch_bounds__(128) void attn_kernel(
    const float* __restrict__ qkv, __bf16* __restrict__ y) {
  __shared__ __bf16 sK[64 * AP];
  __shared__ __bf16 sVT[64 * AP];
  __shared__ __bf16 sP[4 * 16 * AP];
  const int tid = threadIdx.x;
  const int w = tid >> 5;
  const int l = tid & 31;
  const int lr = l & 15, lh = l >> 4;
  const int qt = blockIdx.x, h = blockIdx.y, b = blockIdx.z;
  const int C3 = 3 * CC;
  const float scale = 0.125f;  // 1/sqrt(64)

  // Q fragments (pre-scaled), A-operand swizzle, straight from global
  v16bf qf[2];
  {
    const int row = b * CTT + qt * 64 + w * 16 + lr;
    const float* qp = qkv + (size_t)row * C3 + h * CD;
#pragma unroll
    for (int f = 0; f < 2; ++f) {
      FragBF fr;
#pragma unroll
      for (int i = 0; i < 8; ++i) {
        fr.h[0][i] = (__bf16)(qp[f * 32 + (lh << 3) + i] * scale);
        fr.h[1][i] = (__bf16)(qp[f * 32 + 16 + (lh << 3) + i] * scale);
      }
      qf[f] = fr.v;
    }
  }

  v8f accO[4];
#pragma unroll
  for (int n = 0; n < 4; ++n) accO[n] = vzero8();
  float mrun[8], lrun[8];
#pragma unroll
  for (int v = 0; v < 8; ++v) { mrun[v] = -INFINITY; lrun[v] = 0.f; }

  __bf16* sPw = &sP[w * 16 * AP];

  for (int jt = 0; jt < CTT / 64; ++jt) {
    // stage K tile [key][d] and V tile transposed [d][key], fp32 -> bf16
#pragma unroll
    for (int i = 0; i < 32; ++i) {
      int idx = i * 128 + tid;
      int r = idx >> 6, c = idx & 63;
      size_t g = (size_t)(b * CTT + jt * 64 + r) * C3 + h * CD + c;
      sK[r * AP + c]  = (__bf16)qkv[g + CC];
      sVT[c * AP + r] = (__bf16)qkv[g + 2 * CC];
    }
    __syncthreads();

    // S = Q K^T  (16 x 64 per wave)
    v8f accS[4];
#pragma unroll
    for (int n = 0; n < 4; ++n) accS[n] = vzero8();
#pragma unroll
    for (int f = 0; f < 2; ++f)
#pragma unroll
      for (int n = 0; n < 4; ++n) {
        v16bf kf = ld_fragB(&sK[(n * 16) * AP + f * 32], AP);
        accS[n] = WMMA_BF16(qf[f], kf, accS[n]);
      }

    // mask + online softmax; element (v,l): row M=v+8*lh, col N=lr
    const int irow0 = qt * 64 + w * 16;
#pragma unroll
    for (int v = 0; v < 8; ++v) {
      const int imod = (irow0 + v + (lh << 3)) & 255;
      float sv[4];
      float local = -INFINITY;
#pragma unroll
      for (int n = 0; n < 4; ++n) {
        const int j = jt * 64 + n * 16 + lr;
        float s = accS[n][v];
        if ((j & 255) > imod) s = -INFINITY;
        sv[n] = s;
        local = fmaxf(local, s);
      }
#pragma unroll
      for (int off = 1; off < 16; off <<= 1)
        local = fmaxf(local, __shfl_xor(local, off, 32));
      const float mnew = fmaxf(mrun[v], local);
      const float corr = (mrun[v] == -INFINITY) ? 0.f : __expf(mrun[v] - mnew);
      float psum = 0.f;
#pragma unroll
      for (int n = 0; n < 4; ++n) {
        float p = (sv[n] == -INFINITY || mnew == -INFINITY)
                      ? 0.f : __expf(sv[n] - mnew);
        psum += p;
        sPw[(v + (lh << 3)) * AP + n * 16 + lr] = (__bf16)p;
      }
#pragma unroll
      for (int off = 1; off < 16; off <<= 1)
        psum += __shfl_xor(psum, off, 32);
      lrun[v] = lrun[v] * corr + psum;
      mrun[v] = mnew;
#pragma unroll
      for (int n = 0; n < 4; ++n) accO[n][v] *= corr;
    }
    asm volatile("s_wait_dscnt 0" ::: "memory");  // P LDS turnaround (intra-wave)

    // O += P V
#pragma unroll
    for (int f = 0; f < 2; ++f) {
      v16bf pf = ld_fragA(&sPw[f * 32], AP);
#pragma unroll
      for (int n = 0; n < 4; ++n) {
        v16bf vf = ld_fragB(&sVT[(n * 16) * AP + f * 32], AP);
        accO[n] = WMMA_BF16(pf, vf, accO[n]);
      }
    }
    __syncthreads();
  }

  // normalize + store bf16 into y[B*T, C] at column h*64
#pragma unroll
  for (int v = 0; v < 8; ++v) {
    const float inv = (lrun[v] > 0.f) ? 1.f / lrun[v] : 0.f;
    const int row = b * CTT + qt * 64 + w * 16 + v + (lh << 3);
#pragma unroll
    for (int n = 0; n < 4; ++n)
      y[(size_t)row * CC + h * CD + n * 16 + lr] = (__bf16)(accO[n][v] * inv);
  }
}

// ---- host launcher --------------------------------------------------------
extern "C" void kernel_launch(void* const* d_in, const int* in_sizes, int n_in,
                              void* d_out, int out_size, void* d_ws, size_t ws_size,
                              hipStream_t stream) {
  (void)in_sizes; (void)n_in; (void)out_size; (void)ws_size;
  const int* idx_up    = (const int*)d_in[0];
  const int* idx_down  = (const int*)d_in[1];
  const float* cond    = (const float*)d_in[2];
  const float* tok_up  = (const float*)d_in[3];
  const float* tok_dn  = (const float*)d_in[4];
  const float* pos     = (const float*)d_in[5];
  const float* cond_w  = (const float*)d_in[6];
  const float* cond_b  = (const float*)d_in[7];
  const float* lnf_w   = (const float*)d_in[8];
  const float* lnf_b   = (const float*)d_in[9];
  const float* head_up = (const float*)d_in[10];
  const float* head_dn = (const float*)d_in[11];

  // workspace layout (256B aligned)
  char* w8 = (char*)d_ws;
  size_t off = 0;
  auto bump = [&](size_t bytes) { size_t o = off; off += (bytes + 255) & ~(size_t)255; return o; };
  float*  x    = (float*) (w8 + bump((size_t)CR * CC * 4));
  float*  qkvb = (float*) (w8 + bump((size_t)CR * 3 * CC * 4));
  __bf16* abuf = (__bf16*)(w8 + bump((size_t)CR * CC * 2));
  __bf16* ybuf = (__bf16*)(w8 + bump((size_t)CR * CC * 2));
  __bf16* hbuf = (__bf16*)(w8 + bump((size_t)CR * 4 * CC * 2));

  embed_kernel<<<CR, 256, 0, stream>>>(idx_up, idx_down, cond, tok_up, tok_dn,
                                       pos, cond_w, cond_b, x);

  for (int layer = 0; layer < 12; ++layer) {
    const int base = layer < 6 ? 0 : 10;
    const int lyr = layer % 6;
    const float* ln1    = (const float*)d_in[12 + base] + (size_t)lyr * 2 * CC;
    const float* ln2    = (const float*)d_in[13 + base] + (size_t)lyr * 2 * CC;
    const float* qkv_w  = (const float*)d_in[14 + base] + (size_t)lyr * 3 * CC * CC;
    const float* qkv_b  = (const float*)d_in[15 + base] + (size_t)lyr * 3 * CC;
    const float* proj_w = (const float*)d_in[16 + base] + (size_t)lyr * CC * CC;
    const float* proj_b = (const float*)d_in[17 + base] + (size_t)lyr * CC;
    const float* fc1_w  = (const float*)d_in[18 + base] + (size_t)lyr * 4 * CC * CC;
    const float* fc1_b  = (const float*)d_in[19 + base] + (size_t)lyr * 4 * CC;
    const float* fc2_w  = (const float*)d_in[20 + base] + (size_t)lyr * CC * 4 * CC;
    const float* fc2_b  = (const float*)d_in[21 + base] + (size_t)lyr * CC;

    ln_kernel<<<CR, 256, 0, stream>>>(x, ln1, ln1 + CC, abuf);
    // qkv: bias, f32 out
    gemm_kernel<true, false, false, true, false>
        <<<dim3(3 * CC / 128, CR / 128, 1), 256, 0, stream>>>(
        abuf, CC, 0, qkv_w, qkv_b, qkvb, nullptr, 3 * CC, 0, nullptr, CC);
    attn_kernel<<<dim3(CTT / 64, CH, CB), 128, 0, stream>>>(qkvb, ybuf);
    // proj: bias + residual, f32 out
    gemm_kernel<true, true, false, true, false>
        <<<dim3(CC / 128, CR / 128, 1), 256, 0, stream>>>(
        ybuf, CC, 0, proj_w, proj_b, x, nullptr, CC, 0, x, CC);
    ln_kernel<<<CR, 256, 0, stream>>>(x, ln2, ln2 + CC, abuf);
    // fc1: bias + gelu, bf16 out
    gemm_kernel<true, false, true, false, true>
        <<<dim3(4 * CC / 128, CR / 128, 1), 256, 0, stream>>>(
        abuf, CC, 0, fc1_w, fc1_b, nullptr, hbuf, 4 * CC, 0, nullptr, CC);
    // fc2: bias + residual, f32 out
    gemm_kernel<true, true, false, true, false>
        <<<dim3(CC / 128, CR / 128, 1), 256, 0, stream>>>(
        hbuf, 4 * CC, 0, fc2_w, fc2_b, x, nullptr, CC, 0, x, 4 * CC);
  }

  ln_kernel<<<CR, 256, 0, stream>>>(x, lnf_w, lnf_b, abuf);

  float* outp = (float*)d_out;
  // logits_up: rows t..2t of each batch; batched over B via grid.z
  gemm_kernel<false, false, false, true, false>
      <<<dim3(CV / 128, CT / 128, CB), 256, 0, stream>>>(
      abuf + (size_t)CT * CC, CC, (long long)CTT * CC, head_up, nullptr,
      outp, nullptr, CV, (long long)CT * CV, nullptr, CC);
  // logits_down: rows 2t..3t
  gemm_kernel<false, false, false, true, false>
      <<<dim3(CV / 128, CT / 128, CB), 256, 0, stream>>>(
      abuf + (size_t)2 * CT * CC, CC, (long long)CTT * CC, head_dn, nullptr,
      outp + (size_t)CB * CT * CV, nullptr, CV, (long long)CT * CV, nullptr, CC);
}
